// SelfAttention_43731357007984
// MI455X (gfx1250) — matrix operand order repaired and checked by reading, not compiled, not run on hardware
//
#include <hip/hip_runtime.h>

// ---------------------------------------------------------------------------
// Self-attention (B=4, S=1792, D=1536, NH=12, hd=128) for gfx1250 (MI455X).
// Matmuls via V_WMMA_F32_16X16X32_BF16; shared operand tiles staged through
// LDS with GLOBAL_LOAD_ASYNC_TO_LDS_B128 (double-buffered, s_wait_asynccnt)
// when the toolchain exposes the builtin; synchronous LDS staging otherwise.
// ---------------------------------------------------------------------------

typedef __bf16 bf16;
typedef __bf16 v8bf  __attribute__((ext_vector_type(8)));
typedef __bf16 v16bf __attribute__((ext_vector_type(16)));
typedef float  v8f   __attribute__((ext_vector_type(8)));
typedef int    v4i   __attribute__((ext_vector_type(4)));

#define WMMA_BF16(A, B, C) \
  __builtin_amdgcn_wmma_f32_16x16x32_bf16(false, (A), false, (B), (short)0, (C), false, false)

// ----- async global -> LDS staging (gfx1250 ASYNCcnt path) ------------------
// Builtin signature (from hipcc diagnostics): (v4i __device__*, v4i __shared__*,
// imm offset, imm cpol).
#if defined(__gfx1250__) && __has_builtin(__builtin_amdgcn_global_load_async_to_lds_b128)
#define ASYNC_COPY16(gp, lp)                                           \
  __builtin_amdgcn_global_load_async_to_lds_b128(                      \
      (__attribute__((address_space(1))) v4i*)(void*)(gp),             \
      (__attribute__((address_space(3))) v4i*)(void*)(lp), 0, 0)
#if __has_builtin(__builtin_amdgcn_s_wait_asynccnt)
#define WAIT_ASYNC() __builtin_amdgcn_s_wait_asynccnt(0)
#else
#define WAIT_ASYNC() asm volatile("s_wait_asynccnt 0x0" ::: "memory")
#endif
#else
// Fallback: synchronous 16-byte copy through VGPRs; barriers provide ordering.
#define ASYNC_COPY16(gp, lp) (*(v8bf*)(lp) = *(const v8bf*)(gp))
#define WAIT_ASYNC() ((void)0)
#endif

// ----- fragment loaders -----------------------------------------------------
// A matrix 16x32 (MxK), 16-bit: lane L (r=L&15, hf=L>>4) holds row r,
// K = {hf*8 .. hf*8+7} in VGPRs 0-3 and {16+hf*8 .. 16+hf*8+7} in VGPRs 4-7.
static __device__ __forceinline__ v16bf load_afrag(const bf16* __restrict__ p,
                                                   int ldk, int lane) {
  const int r = lane & 15, hf = lane >> 4;
  const bf16* q = p + (size_t)r * ldk + hf * 8;
  v8bf lo = *(const v8bf*)(q);
  v8bf hi = *(const v8bf*)(q + 16);
  v16bf o;
#pragma unroll
  for (int i = 0; i < 8; ++i) { o[i] = lo[i]; o[i + 8] = hi[i]; }
  return o;
}

// B matrix 32x16 (KxN), 16-bit: lane L holds column n=L&15;
// K = {hf*8 .. hf*8+7} in elems 0-7 and {16+hf*8 .. 16+hf*8+7} in elems 8-15.
static __device__ __forceinline__ v16bf load_bfrag(const bf16* __restrict__ p,
                                                   int ldk, int lane) {
  const int n = lane & 15, hf = lane >> 4;
  const bf16* q = p + (size_t)n * ldk + hf * 8;
  v8bf lo = *(const v8bf*)(q);
  v8bf hi = *(const v8bf*)(q + 16);
  v16bf o;
#pragma unroll
  for (int i = 0; i < 8; ++i) { o[i] = lo[i]; o[i + 8] = hi[i]; }
  return o;
}

// ----- elementwise f32 -> bf16 ---------------------------------------------
__global__ void cvt_f32_to_bf16(const float* __restrict__ in,
                                bf16* __restrict__ out, size_t n) {
  size_t i = (size_t)blockIdx.x * blockDim.x + threadIdx.x;
  const size_t stride = (size_t)gridDim.x * blockDim.x;
  for (; i < n; i += stride) out[i] = (bf16)in[i];
}

// ----- GEMM: C[MxN] (f32) = A[MxK](bf16) * W[NxK](bf16)^T + bias[N] --------
// Block: 256 threads = 8 waves. Tile: 128(M) x 64(N). Wave w -> 16 M-rows,
// 4 N-subtiles. K-step 32. Shared W tile (64x32) staged via async->LDS,
// double buffered; A fragments stream from global (no cross-wave reuse).
__global__ __launch_bounds__(256)
void gemm_bf16_wmma(const bf16* __restrict__ A, const bf16* __restrict__ W,
                    const float* __restrict__ bias, float* __restrict__ C,
                    int M, int N, int K) {
  const int lane = threadIdx.x & 31;
  const int wave = threadIdx.x >> 5;
  const int m0 = blockIdx.x * 128 + wave * 16;
  const int n0 = blockIdx.y * 64;

  __shared__ __attribute__((aligned(64))) bf16 Wt[2][64][32];

  // cooperative staging map: 256 threads x 16B cover the 4KB W tile
  const int tr = threadIdx.x >> 2;         // row 0..63
  const int tc = (threadIdx.x & 3) * 8;    // element offset within row
  const bf16* wsrc = W + (size_t)(n0 + tr) * K + tc;

  ASYNC_COPY16(wsrc, &Wt[0][tr][tc]);      // prologue: kb = 0 -> buf 0

  v8f acc[4] = {};
  const bf16* Abase = A + (size_t)m0 * K;

  for (int kb = 0; kb < K; kb += 32) {
    const int buf = (kb >> 5) & 1;
    WAIT_ASYNC();            // this wave's copies into buf have landed
    __syncthreads();         // everyone's copies visible; prev reads done
    if (kb + 32 < K) {
      ASYNC_COPY16(wsrc + kb + 32, &Wt[buf ^ 1][tr][tc]);
      __builtin_prefetch(Abase + kb + 32, 0, 3);
    }
    v16bf af = load_afrag(Abase + kb, K, lane);
#pragma unroll
    for (int nt = 0; nt < 4; ++nt) {
      v16bf wf = load_bfrag(&Wt[buf][nt * 16][0], 32, lane);
      acc[nt] = WMMA_BF16(af, wf, acc[nt]);
    }
  }

  const int nlane = lane & 15, hf = lane >> 4;
#pragma unroll
  for (int nt = 0; nt < 4; ++nt) {
    const int n = n0 + nt * 16 + nlane;
    const float bv = bias[n];
#pragma unroll
    for (int v = 0; v < 8; ++v) {
      const int m = m0 + v + 8 * hf;
      C[(size_t)m * N + n] = acc[nt][v] + bv;
    }
  }
}

// ----- fused RMSNorm (over D) + RoPE, f32 in -> bf16 out --------------------
__global__ __launch_bounds__(256)
void rmsrope_kernel(const float* __restrict__ Y, const float* __restrict__ g,
                    const float* __restrict__ fcos, const float* __restrict__ fsin,
                    const int* __restrict__ grid_sizes, const int* __restrict__ nmod,
                    bf16* __restrict__ out, int S, int D) {
  const int row = blockIdx.x;        // b*S + s
  const int s = row % S;
  const float* y = Y + (size_t)row * D;

  __shared__ float red[8];
  float ss = 0.f;
  for (int i = threadIdx.x; i < D; i += 256) { float t = y[i]; ss += t * t; }
#pragma unroll
  for (int off = 16; off > 0; off >>= 1) ss += __shfl_xor(ss, off, 32);
  if ((threadIdx.x & 31) == 0) red[threadIdx.x >> 5] = ss;
  __syncthreads();
  float tot = 0.f;
#pragma unroll
  for (int i = 0; i < 8; ++i) tot += red[i];
  const float rn = rsqrtf(tot / (float)D + 1e-6f);

  const int f = grid_sizes[0], hh = grid_sizes[1], ww = grid_sizes[2];
  const int m = nmod[0];
  const int seq_per = f * hh * ww;
  const int total_len = min(seq_per * m, S);
  const bool dorope = (s < total_len);
  const int sp = dorope ? (s % seq_per) : 0;
  const int fi = sp / (hh * ww);
  const int rem = sp % (hh * ww);
  const int hi = rem / ww;
  const int wi = rem % ww;
  const int c = 64, ct = 22, ch = 21;   // c - 2*(c/3), c/3 for c=64

  for (int p = threadIdx.x; p < (D >> 1); p += 256) {
    const int j = p & (c - 1);          // pair index within head
    const int dd0 = 2 * p;              // head*128 + 2*j
    const float yr = y[dd0] * rn * g[dd0];
    const float yi = y[dd0 + 1] * rn * g[dd0 + 1];
    float co = 1.f, si = 0.f;
    if (dorope) {
      const int rowf = (j < ct) ? fi : ((j < ct + ch) ? hi : wi);
      co = fcos[rowf * c + j];
      si = fsin[rowf * c + j];
    }
    out[(size_t)row * D + dd0]     = (bf16)(yr * co - yi * si);
    out[(size_t)row * D + dd0 + 1] = (bf16)(yr * si + yi * co);
  }
}

// ----- flash attention ------------------------------------------------------
// Grid: (S/128, NH, B). Block 256 = 8 waves; wave owns 16 query rows.
// kv tiles of 32; K tile async-staged to LDS (double buffered, shared by all
// waves); V tile transposed into LDS; scores + P*V via WMMA bf16.
#define KVT 32

__global__ __launch_bounds__(256)
void attn_kernel(const bf16* __restrict__ qb, const bf16* __restrict__ kb,
                 const bf16* __restrict__ vb, const int* __restrict__ seq_lens,
                 bf16* __restrict__ ob, int S, int D, int HD) {
  const int lane = threadIdx.x & 31;
  const int wave = threadIdx.x >> 5;
  const int b = blockIdx.z, h = blockIdx.y;
  const int qr0 = blockIdx.x * 128 + wave * 16;
  const int slen = seq_lens[b];
  const float scale = rsqrtf((float)HD);

  __shared__ __attribute__((aligned(64))) bf16 Kt[2][KVT][128];     // K tiles
  __shared__ __attribute__((aligned(64))) bf16 VtS[128 * KVT];      // V^T tile
  __shared__ __attribute__((aligned(64))) bf16 Plds[8][16 * KVT];   // per-wave P

  const int tid = threadIdx.x;
  // K-tile staging map: chunk id = tid*2+c ; row = id>>4 ; elems = (id&15)*8
  auto issueK = [&](int kvbase, int buf) {
#pragma unroll
    for (int cch = 0; cch < 2; ++cch) {
      const int id = tid * 2 + cch;
      const int r = id >> 4;
      const int co = (id & 15) * 8;
      const bf16* gsc = kb + ((size_t)(b * S + kvbase + r) * D + h * HD + co);
      ASYNC_COPY16(gsc, &Kt[buf][r][co]);
    }
  };
  issueK(0, 0);

  // Preload Q fragments for this wave's 16 rows (HD=128 -> 4 K-steps).
  const bf16* qbase = qb + ((size_t)(b * S + qr0) * D + h * HD);
  v16bf qf[4];
#pragma unroll
  for (int kk = 0; kk < 4; ++kk) qf[kk] = load_afrag(qbase + kk * 32, D, lane);

  v8f oacc[8] = {};
  float mrun[8], lrun[8];
#pragma unroll
  for (int v = 0; v < 8; ++v) { mrun[v] = -__builtin_inff(); lrun[v] = 0.f; }

  const int nlane = lane & 15, hf = lane >> 4;

  for (int kv0 = 0; kv0 < S; kv0 += KVT) {
    const int buf = (kv0 / KVT) & 1;
    WAIT_ASYNC();            // this wave's K copies into buf landed
    __syncthreads();         // all waves' copies visible; prev iter reads done

    // --- cooperative V-tile transpose into LDS: VtS[dd][kv] ---
    {
      const int kv = tid >> 3;            // 0..31
      const int dd0 = (tid & 7) * 16;     // 0,16,...,112
      const bf16* vp = vb + ((size_t)(b * S + kv0 + kv) * D + h * HD + dd0);
      v8bf a = *(const v8bf*)(vp);
      v8bf c2 = *(const v8bf*)(vp + 8);
#pragma unroll
      for (int i = 0; i < 8; ++i) {
        VtS[(dd0 + i) * KVT + kv]     = a[i];
        VtS[(dd0 + 8 + i) * KVT + kv] = c2[i];
      }
    }
    if (kv0 + KVT < S) issueK(kv0 + KVT, buf ^ 1);   // overlap with compute
    __syncthreads();

    // --- scores: 16 x 32 tile = 2 n-subtiles x 4 K-steps of WMMA ---
    v8f sfr[2] = {};
#pragma unroll
    for (int kk = 0; kk < 4; ++kk) {
#pragma unroll
      for (int nt = 0; nt < 2; ++nt) {
        v16bf kfrag = load_bfrag(&Kt[buf][nt * 16][kk * 32], 128, lane);
        sfr[nt] = WMMA_BF16(qf[kk], kfrag, sfr[nt]);
      }
    }

    // --- online softmax (rows live in VGPR index, cols in lanes) ---
    float alpha[8];
#pragma unroll
    for (int v = 0; v < 8; ++v) {
      float s0 = sfr[0][v] * scale;
      float s1 = sfr[1][v] * scale;
      if (kv0 + nlane >= slen)      s0 = -1e30f;
      if (kv0 + 16 + nlane >= slen) s1 = -1e30f;
      float mv = fmaxf(s0, s1);
#pragma unroll
      for (int off = 1; off < 16; off <<= 1) mv = fmaxf(mv, __shfl_xor(mv, off, 32));
      const float mnew = fmaxf(mrun[v], mv);
      alpha[v] = __expf(mrun[v] - mnew);
      const float p0 = __expf(s0 - mnew);
      const float p1 = __expf(s1 - mnew);
      sfr[0][v] = p0; sfr[1][v] = p1;
      float ps = p0 + p1;
#pragma unroll
      for (int off = 1; off < 16; off <<= 1) ps += __shfl_xor(ps, off, 32);
      lrun[v] = lrun[v] * alpha[v] + ps;
      mrun[v] = mnew;
    }

    // --- write P (bf16) to wave-private LDS, reload as an A fragment ---
    bf16* pw = &Plds[wave][0];
#pragma unroll
    for (int nt = 0; nt < 2; ++nt)
#pragma unroll
      for (int v = 0; v < 8; ++v)
        pw[(v + 8 * hf) * KVT + nt * 16 + nlane] = (bf16)sfr[nt][v];
    v16bf pf = load_afrag(pw, KVT, lane);

    // --- rescale O and accumulate P*V ---
#pragma unroll
    for (int nt2 = 0; nt2 < 8; ++nt2) {
#pragma unroll
      for (int v = 0; v < 8; ++v) oacc[nt2][v] *= alpha[v];
      v16bf vfrag = load_bfrag(VtS + nt2 * 16 * KVT, KVT, lane);
      oacc[nt2] = WMMA_BF16(pf, vfrag, oacc[nt2]);
    }
  }

  // --- epilogue: O / l, store bf16 in (b,s,h,dd) layout ---
  float inv[8];
#pragma unroll
  for (int v = 0; v < 8; ++v) inv[v] = lrun[v] > 0.f ? 1.f / lrun[v] : 0.f;
#pragma unroll
  for (int nt2 = 0; nt2 < 8; ++nt2) {
    const int dd = nt2 * 16 + nlane;
#pragma unroll
    for (int v = 0; v < 8; ++v) {
      const int r = qr0 + v + 8 * hf;
      ob[(size_t)(b * S + r) * D + h * HD + dd] = (bf16)(oacc[nt2][v] * inv[v]);
    }
  }
}

// ---------------------------------------------------------------------------
extern "C" void kernel_launch(void* const* d_in, const int* in_sizes, int n_in,
                              void* d_out, int out_size, void* d_ws, size_t ws_size,
                              hipStream_t stream) {
  const float* x    = (const float*)d_in[0];
  const int* seq_lens   = (const int*)d_in[1];
  const int* grid_sizes = (const int*)d_in[2];
  const float* fcos = (const float*)d_in[3];
  const float* fsin = (const float*)d_in[4];
  const float* Wq = (const float*)d_in[5];
  const float* bq = (const float*)d_in[6];
  const float* Wk = (const float*)d_in[7];
  const float* bk = (const float*)d_in[8];
  const float* Wv = (const float*)d_in[9];
  const float* bv = (const float*)d_in[10];
  const float* Wo = (const float*)d_in[11];
  const float* bo = (const float*)d_in[12];
  const float* gq = (const float*)d_in[13];
  const float* gk = (const float*)d_in[14];
  const int* nmod = (const int*)d_in[15];

  constexpr int Bc = 4, Sc = 1792, Dc = 1536, NHc = 12, HDc = 128;
  constexpr size_t Mrows = (size_t)Bc * Sc;     // 7168
  constexpr size_t MD = Mrows * Dc;             // 11,010,048 elems
  constexpr size_t WD = (size_t)Dc * Dc;        // 2,359,296 elems

  size_t off = 0;
  auto take = [&](size_t bytes) {
    char* p = (char*)d_ws + off;
    off += (bytes + 255) & ~(size_t)255;
    return (void*)p;
  };
  bf16* xb      = (bf16*)take(MD * 2);
  bf16* wqb     = (bf16*)take(WD * 2);
  bf16* wkb     = (bf16*)take(WD * 2);
  bf16* wvb     = (bf16*)take(WD * 2);
  bf16* wob     = (bf16*)take(WD * 2);
  float* scratch = (float*)take(MD * 4);
  bf16* qb2     = (bf16*)take(MD * 2);
  bf16* kb2     = (bf16*)take(MD * 2);
  bf16* vb2     = (bf16*)take(MD * 2);
  bf16* attnb   = (bf16*)take(MD * 2);
  (void)ws_size; (void)n_in; (void)in_sizes; (void)out_size;

  // 1) convert activations + weights to bf16 (everything stays L2-resident)
  cvt_f32_to_bf16<<<2048, 256, 0, stream>>>(x,  xb,  MD);
  cvt_f32_to_bf16<<<1024, 256, 0, stream>>>(Wq, wqb, WD);
  cvt_f32_to_bf16<<<1024, 256, 0, stream>>>(Wk, wkb, WD);
  cvt_f32_to_bf16<<<1024, 256, 0, stream>>>(Wv, wvb, WD);
  cvt_f32_to_bf16<<<1024, 256, 0, stream>>>(Wo, wob, WD);

  const dim3 ggemm(Mrows / 128, Dc / 64);   // (56, 24)

  // 2) Q = rms(x@Wq^T + bq)*gq, rope
  gemm_bf16_wmma<<<ggemm, 256, 0, stream>>>(xb, wqb, bq, scratch, (int)Mrows, Dc, Dc);
  rmsrope_kernel<<<(int)Mrows, 256, 0, stream>>>(scratch, gq, fcos, fsin, grid_sizes,
                                                 nmod, qb2, Sc, Dc);
  // 3) K
  gemm_bf16_wmma<<<ggemm, 256, 0, stream>>>(xb, wkb, bk, scratch, (int)Mrows, Dc, Dc);
  rmsrope_kernel<<<(int)Mrows, 256, 0, stream>>>(scratch, gk, fcos, fsin, grid_sizes,
                                                 nmod, kb2, Sc, Dc);
  // 4) V (bias added in GEMM; just convert)
  gemm_bf16_wmma<<<ggemm, 256, 0, stream>>>(xb, wvb, bv, scratch, (int)Mrows, Dc, Dc);
  cvt_f32_to_bf16<<<2048, 256, 0, stream>>>(scratch, vb2, MD);

  // 5) attention
  const dim3 gattn(Sc / 128, NHc, Bc);      // (14, 12, 4)
  attn_kernel<<<gattn, 256, 0, stream>>>(qb2, kb2, vb2, seq_lens, attnb, Sc, Dc, HDc);

  // 6) output projection straight into d_out (fp32)
  gemm_bf16_wmma<<<ggemm, 256, 0, stream>>>(attnb, wob, bo, (float*)d_out,
                                            (int)Mrows, Dc, Dc);
}
